// SwinTransformerBlock_6322191860182
// MI455X (gfx1250) — matrix operand order repaired and checked by reading, not compiled, not run on hardware
//
#include <hip/hip_runtime.h>
#include <hip/hip_bf16.h>

// ---------------------------------------------------------------------------
// Swin block for MI455X (gfx1250, wave32).
// GEMMs: v_wmma_f32_16x16x32_bf16, B-tile staged in LDS via
// global_load_async_to_lds_b128 (ASYNCcnt double-buffer), A pipelined in regs.
// Attention (7x7/window/head) is ~1.6 GFLOP -> VALU + wave32 shuffles.
// ---------------------------------------------------------------------------

#define BB 16
#define LL 3584
#define CC 512
#define WW 7
#define SHIFT_ 3
#define NH_ 16
#define HD_ 32
#define HIDDEN_ 2048
#define NWIN_ 512
#define MROWS (BB * LL)   // 57344

typedef __bf16 bf16_t;
typedef __attribute__((ext_vector_type(8)))  bf16_t v8bf;
typedef __attribute__((ext_vector_type(16))) bf16_t v16bf;
typedef __attribute__((ext_vector_type(8)))  float  v8f;

__device__ __forceinline__ unsigned short f32_to_bf16(float f) {
    unsigned u = __float_as_uint(f);
    unsigned r = u + 0x7FFFu + ((u >> 16) & 1u);   // round-to-nearest-even
    return (unsigned short)(r >> 16);
}
__device__ __forceinline__ float bf16_to_f32(unsigned short h) {
    return __uint_as_float(((unsigned)h) << 16);
}

__device__ __forceinline__ v16bf load_frag(const unsigned short* p0,
                                           const unsigned short* p1) {
    v8bf lo = *reinterpret_cast<const v8bf*>(p0);
    v8bf hi = *reinterpret_cast<const v8bf*>(p1);
    return __builtin_shufflevector(lo, hi, 0,1,2,3,4,5,6,7,8,9,10,11,12,13,14,15);
}

// 16B global -> LDS async copy (ASYNCcnt). VDST vgpr = LDS byte address,
// VADDR = 64-bit global address.
__device__ __forceinline__ void async_copy_b128(unsigned lds_addr, const void* gaddr) {
    asm volatile("global_load_async_to_lds_b128 %0, %1, off"
                 :: "v"(lds_addr), "v"(gaddr) : "memory");
}

// ---------------------------------------------------------------------------
// f32 -> bf16 weight conversion
// ---------------------------------------------------------------------------
__global__ void cvt_f32_bf16(const float* __restrict__ in,
                             unsigned short* __restrict__ out, int n) {
    int i = blockIdx.x * 256 + threadIdx.x;
    if (i < n) out[i] = f32_to_bf16(in[i]);
}

// ---------------------------------------------------------------------------
// LayerNorm (+ optional cyclic shift by -SHIFT along L), emit bf16.
// ---------------------------------------------------------------------------
__global__ __launch_bounds__(256)
void ln_bf16_kernel(const float* __restrict__ x, const float* __restrict__ g,
                    const float* __restrict__ b, unsigned short* __restrict__ out,
                    int doShift) {
    int row = blockIdx.x;
    int tid = threadIdx.x;
    int wv = tid >> 5, ln = tid & 31;
    const float* xr = x + (size_t)row * CC;
    float v0 = xr[tid], v1 = xr[tid + 256];

    __shared__ float red[16];
    float s = v0 + v1;
    #pragma unroll
    for (int off = 16; off; off >>= 1) s += __shfl_xor(s, off, 32);
    if (ln == 0) red[wv] = s;
    __syncthreads();
    float mean = 0.f;
    #pragma unroll
    for (int i = 0; i < 8; i++) mean += red[i];
    mean *= (1.0f / CC);

    float d0 = v0 - mean, d1 = v1 - mean;
    float sv = d0 * d0 + d1 * d1;
    #pragma unroll
    for (int off = 16; off; off >>= 1) sv += __shfl_xor(sv, off, 32);
    if (ln == 0) red[8 + wv] = sv;
    __syncthreads();
    float var = 0.f;
    #pragma unroll
    for (int i = 0; i < 8; i++) var += red[8 + i];
    var *= (1.0f / CC);
    float rstd = rsqrtf(var + 1e-5f);

    int dst = row;
    if (doShift) {                       // h' written at (l - SHIFT) mod L
        int bb = row / LL;
        int p  = row - bb * LL;
        int l  = p - SHIFT_; if (l < 0) l += LL;
        dst = bb * LL + l;
    }
    unsigned short* orow = out + (size_t)dst * CC;
    orow[tid]       = f32_to_bf16(d0 * rstd * g[tid]       + b[tid]);
    orow[tid + 256] = f32_to_bf16(d1 * rstd * g[tid + 256] + b[tid + 256]);
}

// ---------------------------------------------------------------------------
// bf16 WMMA GEMM: D[M,N] = A[M,K] @ W[N,K]^T (+bias, + epilogue)
// 256 threads = 8 waves; each wave -> 16(M) x 64(N); block -> 128x64.
// B tile (64x32 bf16 = 4KB) double-buffered in LDS via async-to-LDS copies.
// grid = (N/64, M/128).
// ---------------------------------------------------------------------------
enum { EPI_BF16 = 0, EPI_PROJ = 1, EPI_GELU = 2, EPI_RES = 3 };

template<int EPI>
__global__ __launch_bounds__(256)
void gemm_bf16_wmma(const unsigned short* __restrict__ A,
                    const unsigned short* __restrict__ Wt,
                    const float* __restrict__ bias,
                    const float* __restrict__ res,
                    float* __restrict__ outF,
                    unsigned short* __restrict__ outB,
                    int M, int N, int K) {
    __shared__ unsigned short bsh[2][64 * 32];   // [buf][n*32 + k]

    const int tid  = threadIdx.x;
    const int lane = tid & 31;
    const int wave = tid >> 5;
    const int half = lane >> 4;         // lane[4]
    const int l16  = lane & 15;

    const int m0 = blockIdx.y * 128 + wave * 16;
    const int n0 = blockIdx.x * 64;

    // --- B staging coords: thread copies 16B: row sn, k-slice sq*8 ---
    const int sn = tid >> 2;            // 0..63
    const int sq = tid & 3;             // 0..3
    const unsigned short* gB = Wt + (size_t)(n0 + sn) * K + sq * 8;
    const unsigned ldsB0 = (unsigned)(uintptr_t)&bsh[0][sn * 32 + sq * 8];
    const unsigned ldsB1 = (unsigned)(uintptr_t)&bsh[1][sn * 32 + sq * 8];

    // --- A fragment (16x32 bf16): lanes 0-15 -> K {0..7,16..23}, 16-31 -> {8..15,24..31}
    const unsigned short* arow = A + (size_t)(m0 + l16) * K + half * 8;

    const int T = K >> 5;               // k-tiles of 32

    // prologue: stage tiles 0 and 1
    async_copy_b128(ldsB0, gB);
    async_copy_b128(ldsB1, gB + 32);

    v8f acc[4] = {};
    v16bf a = load_frag(arow, arow + 16);

    for (int t = 0; t < T; ++t) {
        // tile t resident once our (in-order) asyncs beyond it are the only ones left
        if (t + 1 < T) asm volatile("s_wait_asynccnt 0x1" ::: "memory");
        else           asm volatile("s_wait_asynccnt 0x0" ::: "memory");
        __syncthreads();

        // register-pipeline next A fragment
        v16bf an = a;
        if (t + 1 < T) {
            const unsigned short* ap = arow + (size_t)(t + 1) * 32;
            __builtin_prefetch(ap + 128, 0, 1);          // global_prefetch_b8
            an = load_frag(ap, ap + 16);
        }

        const unsigned short* bb = &bsh[t & 1][0];
        #pragma unroll
        for (int tt = 0; tt < 4; ++tt) {
            // B frag (32x16): lane = col n, half 0 -> K {0..7,16..23}, half 1 -> K {8..15,24..31}
            const unsigned short* bp = bb + (16 * tt + l16) * 32 + half * 8;
            v16bf b = load_frag(bp, bp + 16);
            acc[tt] = __builtin_amdgcn_wmma_f32_16x16x32_bf16(
                false, a, false, b, (short)0, acc[tt], false, false);
        }

        __syncthreads();                 // all waves done with buf[t&1]
        if (t + 2 < T)                   // tile t+2 -> buf[(t+2)&1] == buf[t&1]
            async_copy_b128((t & 1) ? ldsB1 : ldsB0, gB + (size_t)(t + 2) * 32);
        a = an;
    }

    // C/D layout: VGPR i -> M = m0 + i + half*8, N = n0 + 16tt + l16
    #pragma unroll
    for (int tt = 0; tt < 4; ++tt) {
        int col = n0 + 16 * tt + l16;
        float bv = bias[col];
        #pragma unroll
        for (int i = 0; i < 8; i++) {
            int row = m0 + half * 8 + i;
            float val = acc[tt][i] + bv;
            if constexpr (EPI == EPI_BF16) {
                outB[(size_t)row * N + col] = f32_to_bf16(val);
            } else if constexpr (EPI == EPI_GELU) {
                float gl = 0.5f * val * (1.0f + erff(val * 0.70710678118654752f));
                outB[(size_t)row * N + col] = f32_to_bf16(gl);
            } else if constexpr (EPI == EPI_PROJ) {
                // window-reverse + roll(+SHIFT): out_final[(p+SHIFT)%L] = out[p]
                int bb2 = row / LL;
                int p   = row - bb2 * LL;
                int l   = p + SHIFT_; if (l >= LL) l -= LL;
                size_t di = (size_t)(bb2 * LL + l) * N + col;
                outF[di] = res[di] + val;
            } else { // EPI_RES
                size_t di = (size_t)row * N + col;
                outF[di] = res[di] + val;
            }
        }
    }
}

// ---------------------------------------------------------------------------
// Windowed attention: one block per window (8192), one wave per head (16).
// ---------------------------------------------------------------------------
__global__ __launch_bounds__(512)
void attn_kernel(const unsigned short* __restrict__ qkv,   // (M, 3C) bf16
                 const float* __restrict__ rel_table,      // (169, NH)
                 unsigned short* __restrict__ out) {       // (M, C) bf16
    const int win  = blockIdx.x;            // 0 .. B*NWIN-1
    const int head = threadIdx.x >> 5;      // 0..15
    const int lane = threadIdx.x & 31;      // d
    const int wn   = win % NWIN_;           // window-in-batch
    const float scale = 0.17677669529663689f;   // HD^-0.5

    const size_t base = (size_t)win * WW * (3 * CC);
    float q[WW], k[WW], v[WW];
    #pragma unroll
    for (int i = 0; i < WW; i++) {
        const unsigned short* r = qkv + base + (size_t)i * (3 * CC) + head * HD_ + lane;
        q[i] = bf16_to_f32(r[0]) * scale;
        k[i] = bf16_to_f32(r[CC]);
        v[i] = bf16_to_f32(r[2 * CC]);
    }

    const bool masked = (wn == NWIN_ - 1);  // only last window has shift mask
    float s[WW][WW];
    #pragma unroll
    for (int i = 0; i < WW; i++) {
        #pragma unroll
        for (int j = 0; j < WW; j++) {
            float p = q[i] * k[j];
            #pragma unroll
            for (int off = 16; off; off >>= 1) p += __shfl_xor(p, off, 32);
            float bias = rel_table[(84 + i - j) * NH_ + head]; // idx = 84 + i - j
            float mv = (masked && ((i < WW - SHIFT_) != (j < WW - SHIFT_))) ? -100.0f : 0.0f;
            s[i][j] = p + bias + mv;
        }
    }

    #pragma unroll
    for (int i = 0; i < WW; i++) {
        float mx = s[i][0];
        #pragma unroll
        for (int j = 1; j < WW; j++) mx = fmaxf(mx, s[i][j]);
        float sum = 0.f;
        #pragma unroll
        for (int j = 0; j < WW; j++) { float e = __expf(s[i][j] - mx); s[i][j] = e; sum += e; }
        float inv = 1.0f / sum;
        float o = 0.f;
        #pragma unroll
        for (int j = 0; j < WW; j++) o += s[i][j] * v[j];
        o *= inv;
        out[(size_t)(win * WW + i) * CC + head * HD_ + lane] = f32_to_bf16(o);
    }
}

// ---------------------------------------------------------------------------
// Launch. Inputs (setup_inputs order):
// 0:x 1:g1 2:b1 3:qkv_w 4:qkv_b 5:rel_table 6:proj_w 7:proj_b 8:g2 9:b2
// 10:fc1_w 11:fc1_b 12:fc2_w 13:fc2_b
// Workspace (~418 MB):
//  [wbf weights bf16 | h_bf (reused for h2) | x1 f32 | BIG: qkv+attnout -> hidden]
// ---------------------------------------------------------------------------
extern "C" void kernel_launch(void* const* d_in, const int* in_sizes, int n_in,
                              void* d_out, int out_size, void* d_ws, size_t ws_size,
                              hipStream_t stream) {
    const float* x       = (const float*)d_in[0];
    const float* g1      = (const float*)d_in[1];
    const float* b1      = (const float*)d_in[2];
    const float* qkv_w   = (const float*)d_in[3];
    const float* qkv_b   = (const float*)d_in[4];
    const float* rel_tab = (const float*)d_in[5];
    const float* proj_w  = (const float*)d_in[6];
    const float* proj_b  = (const float*)d_in[7];
    const float* g2      = (const float*)d_in[8];
    const float* b2      = (const float*)d_in[9];
    const float* fc1_w   = (const float*)d_in[10];
    const float* fc1_b   = (const float*)d_in[11];
    const float* fc2_w   = (const float*)d_in[12];
    const float* fc2_b   = (const float*)d_in[13];
    float* out = (float*)d_out;

    char* ws = (char*)d_ws;
    const size_t N_QKVW = (size_t)3 * CC * CC;     //  786432
    const size_t N_PRJW = (size_t)CC * CC;         //  262144
    const size_t N_FC1W = (size_t)HIDDEN_ * CC;    // 1048576
    const size_t N_FC2W = (size_t)CC * HIDDEN_;    // 1048576

    size_t o = 0;
    unsigned short* w_qkv = (unsigned short*)(ws + o); o += N_QKVW * 2;
    unsigned short* w_prj = (unsigned short*)(ws + o); o += N_PRJW * 2;
    unsigned short* w_fc1 = (unsigned short*)(ws + o); o += N_FC1W * 2;
    unsigned short* w_fc2 = (unsigned short*)(ws + o); o += N_FC2W * 2;
    unsigned short* h_bf  = (unsigned short*)(ws + o); o += (size_t)MROWS * CC * 2;
    float*          x1    = (float*)(ws + o);          o += (size_t)MROWS * CC * 4;
    char*           big   = ws + o;
    unsigned short* qkv_bf  = (unsigned short*)big;                                // M x 1536
    unsigned short* attn_bf = (unsigned short*)(big + (size_t)MROWS * 3 * CC * 2); // M x 512
    unsigned short* hid_bf  = (unsigned short*)big;   // reuses qkv+attn region, M x 2048

    // 1) weights -> bf16
    cvt_f32_bf16<<<dim3((unsigned)((N_QKVW + 255) / 256)), 256, 0, stream>>>(qkv_w, w_qkv, (int)N_QKVW);
    cvt_f32_bf16<<<dim3((unsigned)((N_PRJW + 255) / 256)), 256, 0, stream>>>(proj_w, w_prj, (int)N_PRJW);
    cvt_f32_bf16<<<dim3((unsigned)((N_FC1W + 255) / 256)), 256, 0, stream>>>(fc1_w, w_fc1, (int)N_FC1W);
    cvt_f32_bf16<<<dim3((unsigned)((N_FC2W + 255) / 256)), 256, 0, stream>>>(fc2_w, w_fc2, (int)N_FC2W);

    // 2) LN1 + roll(-SHIFT) -> bf16
    ln_bf16_kernel<<<dim3(MROWS), 256, 0, stream>>>(x, g1, b1, h_bf, 1);

    // 3) QKV GEMM: (57344 x 512) @ (512 x 1536) + bias -> bf16
    gemm_bf16_wmma<EPI_BF16><<<dim3(3 * CC / 64, MROWS / 128), 256, 0, stream>>>(
        h_bf, w_qkv, qkv_b, nullptr, nullptr, qkv_bf, MROWS, 3 * CC, CC);

    // 4) windowed attention -> bf16
    attn_kernel<<<dim3(BB * NWIN_), 512, 0, stream>>>(qkv_bf, rel_tab, attn_bf);

    // 5) proj GEMM + un-shift + residual(x) -> x1 (f32)
    gemm_bf16_wmma<EPI_PROJ><<<dim3(CC / 64, MROWS / 128), 256, 0, stream>>>(
        attn_bf, w_prj, proj_b, x, x1, nullptr, MROWS, CC, CC);

    // 6) LN2 -> bf16 (reuse h_bf)
    ln_bf16_kernel<<<dim3(MROWS), 256, 0, stream>>>(x1, g2, b2, h_bf, 0);

    // 7) FC1 GEMM + exact GELU -> bf16 (hidden reuses BIG region)
    gemm_bf16_wmma<EPI_GELU><<<dim3(HIDDEN_ / 64, MROWS / 128), 256, 0, stream>>>(
        h_bf, w_fc1, fc1_b, nullptr, nullptr, hid_bf, MROWS, HIDDEN_, CC);

    // 8) FC2 GEMM + residual(x1) -> d_out (f32)
    gemm_bf16_wmma<EPI_RES><<<dim3(CC / 64, MROWS / 128), 256, 0, stream>>>(
        hid_bf, w_fc2, fc2_b, x1, out, nullptr, MROWS, CC, HIDDEN_);
}
